// MOELayer_51659866636789
// MI455X (gfx1250) — matrix operand order repaired and checked by reading, not compiled, not run on hardware
//
#include <hip/hip_runtime.h>
#include <hip/hip_bf16.h>
#include <math.h>

// ---------------- problem constants (fixed by the reference) ----------------
#define NTOK 2048   // B*T = 4*512
#define DIM  1024   // D
#define HID  4096   // H = 4*D
#define NEXP 8
#define CAP  640    // int(2 * 1.25 * 2048 / 8), already even

typedef __attribute__((ext_vector_type(16))) __bf16          v16bf;
typedef __attribute__((ext_vector_type(8)))  float           v8f;
typedef __attribute__((ext_vector_type(8)))  unsigned short  us8;

union FragCvt { us8 h[2]; v16bf v; };   // two 16B halves -> one bf16 fragment

static __device__ __forceinline__ unsigned short f2bf(float f) {
    unsigned int u = __float_as_uint(f);
    unsigned int r = u + 0x7FFFu + ((u >> 16) & 1u);   // round-to-nearest-even
    return (unsigned short)(r >> 16);
}

// ---------------------------- router (top-2 gate) ---------------------------
__global__ void router_kernel(const float* __restrict__ x,
                              const float* __restrict__ wg,
                              int* __restrict__ topi,
                              float* __restrict__ wts_raw) {
    int n = blockIdx.x * blockDim.x + threadIdx.x;
    if (n >= NTOK) return;
    const float* xr = x + (size_t)n * DIM;
    float acc[NEXP];
#pragma unroll
    for (int e = 0; e < NEXP; ++e) acc[e] = 0.f;
    for (int d = 0; d < DIM; ++d) {
        float xv = xr[d];
#pragma unroll
        for (int e = 0; e < NEXP; ++e) acc[e] += xv * wg[d * NEXP + e];
    }
    int i1 = 0; float v1 = acc[0];
#pragma unroll
    for (int e = 1; e < NEXP; ++e) if (acc[e] > v1) { v1 = acc[e]; i1 = e; }
    int i2 = -1; float v2 = -3.0e38f;
#pragma unroll
    for (int e = 0; e < NEXP; ++e) {
        if (e == i1) continue;
        if (acc[e] > v2) { v2 = acc[e]; i2 = e; }
    }
    float ex = expf(v2 - v1);           // v1 >= v2
    float p1 = 1.0f / (1.0f + ex);
    float p2 = ex * p1;
    topi[n * 2 + 0] = i1;
    topi[n * 2 + 1] = i2;
    wts_raw[n * 2 + 0] = p1;
    wts_raw[n * 2 + 1] = p2;
}

// ------- per-expert arrival rank (k-major order), capacity drop -------------
__global__ void rank_kernel(const int* __restrict__ topi,
                            const float* __restrict__ wts_raw,
                            float* __restrict__ wts2,   // [2][NTOK]
                            int* __restrict__ slot) {   // [2][NTOK]
    int e = threadIdx.x;
    if (e >= NEXP) return;
    int cnt = 0;
    for (int a = 0; a < 2 * NTOK; ++a) {
        int k = (a >= NTOK) ? 1 : 0;
        int n = a - k * NTOK;
        if (topi[n * 2 + k] == e) {
            bool keep = (cnt < CAP);
            slot[k * NTOK + n] = keep ? cnt : 0;
            wts2[k * NTOK + n] = keep ? wts_raw[n * 2 + k] : 0.f;
            ++cnt;
        }
    }
}

// ------------------------------ zero fill -----------------------------------
__global__ void fill_zero_u32(unsigned int* __restrict__ p, unsigned int n32) {
    unsigned int i = blockIdx.x * blockDim.x + threadIdx.x;
    if (i < n32) p[i] = 0u;
}

// --------------------------- dispatch (scatter) -----------------------------
__global__ void dispatch_kernel(const float* __restrict__ x,
                                const int* __restrict__ topi,
                                const float* __restrict__ wts2,
                                const int* __restrict__ slot,
                                unsigned short* __restrict__ exp_x) { // bf16 [E][CAP][DIM]
    int a = blockIdx.x;                      // 0 .. 2*NTOK-1
    int k = (a >= NTOK) ? 1 : 0;
    int n = a - k * NTOK;
    float w = wts2[k * NTOK + n];
    if (w == 0.f) return;                    // dropped or zero gate (cb==0)
    int e = topi[n * 2 + k];
    int c = slot[k * NTOK + n];
    unsigned short* dst = exp_x + ((size_t)e * CAP + c) * DIM;
    const float* src = x + (size_t)n * DIM;
    for (int d = threadIdx.x; d < DIM; d += blockDim.x) dst[d] = f2bf(src[d]);
}

// ----------------- grouped GEMM: bf16 WMMA, f32 accumulate ------------------
// A: bf16 [E][M][K] row-major; B: f32 [E][K][N] row-major (converted on the fly)
// Workgroup tile 64(M) x 128(N), 8 wave32s in a 2x4 grid, 2x2 WMMA tiles/wave.
// A tile staged with async global->LDS copies (ASYNCcnt); B tile converted
// f32->bf16 in-register and stored transposed so fragments are ds_load_b128.
// MODE 0: out = bf16(gelu_exact(acc)) -> outBf ; MODE 1: out = acc (f32) -> outF
template <int MODE, int M, int K, int N>
__global__ __launch_bounds__(256)
void moe_gemm(const unsigned short* __restrict__ A,
              const float* __restrict__ B,
              unsigned short* __restrict__ outBf,
              float* __restrict__ outF) {
    __shared__ __align__(16) unsigned short As[64][40];    // 64 M x 32 K (+pad)
    __shared__ __align__(16) unsigned short Bs[128][40];   // 128 N x 32 K (+pad)

    const int e = blockIdx.z;
    A += (size_t)e * M * K;
    B += (size_t)e * K * N;
    const size_t obase = (size_t)e * M * N;

    const int tid  = threadIdx.x;
    const int lane = tid & 31;
    const int wave = tid >> 5;          // 8 wave32s
    const int wm   = wave >> 2;         // 0..1 : 32-row band
    const int wn   = wave & 3;          // 0..3 : 32-col band
    const int m0   = blockIdx.y * 64;
    const int n0   = blockIdx.x * 128;

    const int lm   = lane & 15;
    const int akb  = (lane & 16) ? 8 : 0;    // A 16x32 frag: lane K-half select
    const int bkb  = (lane & 16) ? 8 : 0;    // B 32x16 frag: lane K-half select

    // ---- A async-copy mapping: each thread owns one 16B chunk of the tile --
    const int ar_row = tid >> 2;             // 0..63 (row), 4 chunks per row
    const int akp    = tid & 3;              // 16B chunk index within row
    // LDS byte offset = low 32 bits of the generic (flat) shared address
    const unsigned lds_a =
        (unsigned)(unsigned long long)(const void*)&As[ar_row][akp * 8];
    const unsigned short* ga_row = A + (size_t)(m0 + ar_row) * K + akp * 8;

    // ---- B staging mapping: each thread owns one N column half (16 K vals) -
    const int bnc = tid & 127;          // column 0..127
    const int bkh = (tid >> 7) * 16;    // K offset 0 or 16

    v8f acc[2][2] = {};

    for (int kk = 0; kk < K; kk += 32) {
        // --- stage A: one async global->LDS 16B copy per thread ---
        {
            unsigned long long gsrc = (unsigned long long)(const void*)(ga_row + kk);
            asm volatile("global_load_async_to_lds_b128 %0, %1, off"
                         :: "v"(lds_a), "v"(gsrc) : "memory");
        }
        // --- stage B (f32 -> bf16, transposed), overlaps the async copy ---
        {
            const float* gb = B + (size_t)(kk + bkh) * N + n0 + bnc;
#pragma unroll
            for (int h = 0; h < 2; ++h) {
                us8 pk;
#pragma unroll
                for (int j = 0; j < 8; ++j)
                    pk[j] = f2bf(gb[(size_t)(h * 8 + j) * N]);
                *(us8*)&Bs[bnc][bkh + h * 8] = pk;   // 16B aligned (row=80B)
            }
        }
        // prefetch next B K-chunk while this one is consumed
        if (kk + 32 < K) {
            if (tid < 32)
                __builtin_prefetch(B + (size_t)(kk + 32 + tid) * N + n0, 0, 0);
        }
        // async A copy must land in LDS before the tile barrier
        asm volatile("s_wait_asynccnt 0x0" ::: "memory");
        __syncthreads();

        // --- fragments: 2x ds_load_b128 each, per ISA 7.12.2 layouts ---
        FragCvt fa[2], fb[2];
#pragma unroll
        for (int mt = 0; mt < 2; ++mt) {
            const unsigned short* ap = &As[wm * 32 + mt * 16 + lm][0];
            fa[mt].h[0] = *(const us8*)(ap + akb);        // elems 0..7 : K=akb..
            fa[mt].h[1] = *(const us8*)(ap + 16 + akb);   // elems 8..15: K=16+akb..
        }
#pragma unroll
        for (int nt = 0; nt < 2; ++nt) {
            const unsigned short* bp = &Bs[wn * 32 + nt * 16 + lm][0];
            fb[nt].h[0] = *(const us8*)(bp + bkb);        // elems 0..7 : K=bkb..
            fb[nt].h[1] = *(const us8*)(bp + 16 + bkb);   // elems 8..15: K=16+bkb..
        }
#pragma unroll
        for (int mt = 0; mt < 2; ++mt)
#pragma unroll
            for (int nt = 0; nt < 2; ++nt)
                acc[mt][nt] = __builtin_amdgcn_wmma_f32_16x16x32_bf16(
                    false, fa[mt].v, false, fb[nt].v, (short)0,
                    acc[mt][nt], false, false);

        __syncthreads();
    }

    // --- epilogue: C/D layout lanes 0-15 rows 0-7, lanes 16-31 rows 8-15 ---
    const int rlo = (lane & 16) ? 8 : 0;
#pragma unroll
    for (int mt = 0; mt < 2; ++mt) {
#pragma unroll
        for (int nt = 0; nt < 2; ++nt) {
            const int row0 = m0 + wm * 32 + mt * 16 + rlo;
            const int col  = n0 + wn * 32 + nt * 16 + lm;
#pragma unroll
            for (int v = 0; v < 8; ++v) {
                float val = acc[mt][nt][v];
                if (MODE == 0) {
                    float g = 0.5f * val * (1.0f + erff(val * 0.7071067811865475f));
                    outBf[obase + (size_t)(row0 + v) * N + col] = f2bf(g);
                } else {
                    outF[obase + (size_t)(row0 + v) * N + col] = val;
                }
            }
        }
    }
}

// ------------------------------- combine ------------------------------------
__global__ void combine_kernel(const float* __restrict__ exp_out, // [E][CAP][DIM]
                               const int* __restrict__ topi,
                               const float* __restrict__ wts2,
                               const int* __restrict__ slot,
                               float* __restrict__ out) {
    int n = blockIdx.x;
    int e0 = topi[n * 2 + 0], e1 = topi[n * 2 + 1];
    float w0 = wts2[0 * NTOK + n], w1 = wts2[1 * NTOK + n];
    int c0 = slot[0 * NTOK + n], c1 = slot[1 * NTOK + n];
    const float* p0 = exp_out + ((size_t)e0 * CAP + c0) * DIM;
    const float* p1 = exp_out + ((size_t)e1 * CAP + c1) * DIM;
    for (int d = threadIdx.x; d < DIM; d += blockDim.x)
        out[(size_t)n * DIM + d] = w0 * p0[d] + w1 * p1[d];
}

// ------------------------------- launcher -----------------------------------
extern "C" void kernel_launch(void* const* d_in, const int* in_sizes, int n_in,
                              void* d_out, int out_size, void* d_ws, size_t ws_size,
                              hipStream_t stream) {
    const float* x     = (const float*)d_in[0];   // [N, D]  (flattened B*T)
    const float* wg    = (const float*)d_in[1];   // [D, 8]
    const float* cfc   = (const float*)d_in[2];   // [8, D, H]
    const float* cproj = (const float*)d_in[3];   // [8, H, D]
    float* out = (float*)d_out;

    // workspace layout (256B-aligned slabs)
    char* ws = (char*)d_ws;
    size_t off = 0;
    auto take = [&](size_t bytes) { char* p = ws + off; off = (off + bytes + 255) & ~(size_t)255; return p; };
    int*            topi    = (int*)take((size_t)2 * NTOK * 4);
    float*          wts_raw = (float*)take((size_t)2 * NTOK * 4);
    float*          wts2    = (float*)take((size_t)2 * NTOK * 4);
    int*            slot    = (int*)take((size_t)2 * NTOK * 4);
    unsigned short* exp_x   = (unsigned short*)take((size_t)NEXP * CAP * DIM * 2); // bf16
    unsigned short* hbuf    = (unsigned short*)take((size_t)NEXP * CAP * HID * 2); // bf16
    float*          exp_out = (float*)take((size_t)NEXP * CAP * DIM * 4);
    (void)in_sizes; (void)n_in; (void)out_size; (void)ws_size;

    // 1) router
    router_kernel<<<NTOK / 256, 256, 0, stream>>>(x, wg, topi, wts_raw);
    // 2) capacity ranking (sequential per expert, matches reference cumsum order)
    rank_kernel<<<1, 32, 0, stream>>>(topi, wts_raw, wts2, slot);
    // 3) zero expert input buffer, then scatter tokens
    {
        unsigned int n32 = (unsigned int)((size_t)NEXP * CAP * DIM / 2);
        fill_zero_u32<<<(n32 + 255) / 256, 256, 0, stream>>>((unsigned int*)exp_x, n32);
    }
    dispatch_kernel<<<2 * NTOK, 256, 0, stream>>>(x, topi, wts2, slot, exp_x);
    // 4) expert GEMM 1: h = gelu(exp_x @ c_fc)   [E, CAP, HID] bf16
    {
        dim3 grid(HID / 128, CAP / 64, NEXP);
        moe_gemm<0, CAP, DIM, HID><<<grid, 256, 0, stream>>>(exp_x, cfc, hbuf, nullptr);
    }
    // 5) expert GEMM 2: exp_out = h @ c_proj     [E, CAP, DIM] f32
    {
        dim3 grid(DIM / 128, CAP / 64, NEXP);
        moe_gemm<1, CAP, HID, DIM><<<grid, 256, 0, stream>>>(hbuf, cproj, nullptr, exp_out);
    }
    // 6) weighted combine
    combine_kernel<<<NTOK, 256, 0, stream>>>(exp_out, topi, wts2, slot, out);
}
